// LLaDA2MoeDecoderLayer_506806141423
// MI455X (gfx1250) — matrix operand order repaired and checked by reading, not compiled, not run on hardware
//
#include <hip/hip_runtime.h>
#include <cstdint>
#include <cstddef>

#ifndef __has_builtin
#define __has_builtin(x) 0
#endif

#if defined(__HIP_DEVICE_COMPILE__) && \
    __has_builtin(__builtin_amdgcn_global_load_async_to_lds_b128)
#define HAVE_ASYNC_LDS 1
#else
#define HAVE_ASYNC_LDS 0
#endif

// ---------------------------------------------------------------------------
// CDNA5 (gfx1250) wave32 WMMA types
// ---------------------------------------------------------------------------
typedef __attribute__((ext_vector_type(16))) __bf16 v16bf;
typedef __attribute__((ext_vector_type(8)))  __bf16 v8bf;
typedef __attribute__((ext_vector_type(8)))  float  v8f;
typedef __attribute__((ext_vector_type(4)))  int    v4i;

#if HAVE_ASYNC_LDS
// The async-LDS builtin takes int4 pointers qualified with explicit address
// spaces (AS1 = global/"__device__", AS3 = LDS). The qualifiers must live on
// a typedef so C-style casts actually carry them.
typedef __attribute__((address_space(1))) v4i as1_v4i;
typedef __attribute__((address_space(3))) v4i as3_v4i;
#endif

union ABFrag { v16bf v; v8bf h[2]; };

__device__ __forceinline__ v8f wmma_bf16(const ABFrag& a, const ABFrag& b, v8f c) {
  // D = A(16x32 bf16) * B(32x16 bf16) + C(16x16 f32)
  return __builtin_amdgcn_wmma_f32_16x16x32_bf16(false, a.v, false, b.v,
                                                 (short)0, c, false, false);
}

__device__ __forceinline__ v8f zero_v8f() {
  v8f z;
#pragma unroll
  for (int i = 0; i < 8; ++i) z[i] = 0.f;
  return z;
}

// 16-byte global -> LDS copy. On CDNA5 use the ASYNCcnt-tracked async path
// (GLOBAL_LOAD_ASYNC_TO_LDS_B128): data lands in LDS without passing through
// VGPRs, overlapping with WMMA compute. Fallback: synchronous vector copy.
__device__ __forceinline__ void async_copy16(const __bf16* g, __bf16* l) {
#if HAVE_ASYNC_LDS
  __builtin_amdgcn_global_load_async_to_lds_b128((as1_v4i*)g, (as3_v4i*)l, 0, 0);
#else
  *(v8bf*)l = *(const v8bf*)g;
#endif
}

__device__ __forceinline__ void async_wait0() {
#if HAVE_ASYNC_LDS
#if __has_builtin(__builtin_amdgcn_s_wait_asynccnt)
  __builtin_amdgcn_s_wait_asynccnt(0);
#else
  asm volatile("s_wait_asynccnt 0x0" ::: "memory");
#endif
#endif
}

// ---------------------------------------------------------------------------
// Generic bf16 WMMA GEMM:  C[M][N] (f32) = A[M][K] (bf16, row-major)
//                                        @ Bt[N][K] (bf16, pre-transposed)
// Block tile 128x128, 8 waves (4x2), each wave 32x64 (2x4 16x16 frags).
// K step 32, double-buffered LDS fed by async global->LDS copies.
// EPI: 0 = store, 1 = store + residual, 2 = C += rowscale(row) * acc
// ---------------------------------------------------------------------------
template <int EPI>
__global__ __launch_bounds__(256) void gemm_bf16_wmma(
    const __bf16* __restrict__ A, const __bf16* __restrict__ Bt,
    float* __restrict__ C, int M, int N, int K,
    const float* __restrict__ res, const float* __restrict__ rowscale,
    int rsStride) {
  __shared__ __align__(16) __bf16 As[2][128][40];
  __shared__ __align__(16) __bf16 Bs[2][128][40];

  const int tid  = threadIdx.x;
  const int lane = tid & 31;
  const int wave = tid >> 5;
  const int hi   = lane >> 4;
  const int ln   = lane & 15;
  const int wm   = wave & 3;   // 4 waves along M (32 rows each)
  const int wn   = wave >> 2;  // 2 waves along N (64 cols each)
  const int m0   = blockIdx.y * 128;
  const int n0   = blockIdx.x * 128;
  const int lr   = tid >> 1;          // 0..127: tile row loaded by this thread
  const int lc   = (tid & 1) * 16;    // 0 or 16: half-offset within 32-K slab

  const __bf16* Arow = A  + (size_t)(m0 + lr) * K + lc;
  const __bf16* Brow = Bt + (size_t)(n0 + lr) * K + lc;

  v8f acc[2][4];
#pragma unroll
  for (int i = 0; i < 2; ++i)
#pragma unroll
    for (int j = 0; j < 4; ++j) acc[i][j] = zero_v8f();

  // prologue: stage K-slab 0 into buffer 0
  async_copy16(Arow,     &As[0][lr][lc]);
  async_copy16(Arow + 8, &As[0][lr][lc + 8]);
  async_copy16(Brow,     &Bs[0][lr][lc]);
  async_copy16(Brow + 8, &Bs[0][lr][lc + 8]);
  async_wait0();
  __syncthreads();

  const int nk = K >> 5;
  for (int kt = 0; kt < nk; ++kt) {
    const int cur = kt & 1;
    if (kt + 1 < nk) {  // prefetch next K-slab into the other buffer
      const int k1 = (kt + 1) << 5;
      async_copy16(Arow + k1,     &As[cur ^ 1][lr][lc]);
      async_copy16(Arow + k1 + 8, &As[cur ^ 1][lr][lc + 8]);
      async_copy16(Brow + k1,     &Bs[cur ^ 1][lr][lc]);
      async_copy16(Brow + k1 + 8, &Bs[cur ^ 1][lr][lc + 8]);
    }

    ABFrag a[2], b[4];
#pragma unroll
    for (int i = 0; i < 2; ++i) {
      const int row = wm * 32 + i * 16 + ln;             // A lane = M row
      a[i].h[0] = *(const v8bf*)&As[cur][row][8 * hi];       // K = 8*hi..
      a[i].h[1] = *(const v8bf*)&As[cur][row][16 + 8 * hi];  // K = 16+8*hi..
    }
#pragma unroll
    for (int j = 0; j < 4; ++j) {
      const int col = wn * 64 + j * 16 + ln;             // B lane = N col
      b[j].h[0] = *(const v8bf*)&Bs[cur][col][8 * hi];       // K = 8*hi..
      b[j].h[1] = *(const v8bf*)&Bs[cur][col][16 + 8 * hi];
    }
#pragma unroll
    for (int i = 0; i < 2; ++i)
#pragma unroll
      for (int j = 0; j < 4; ++j) acc[i][j] = wmma_bf16(a[i], b[j], acc[i][j]);

    async_wait0();     // next slab fully landed in LDS
    __syncthreads();   // all waves done reading cur + next slab visible
  }

#pragma unroll
  for (int i = 0; i < 2; ++i)
#pragma unroll
    for (int j = 0; j < 4; ++j)
#pragma unroll
      for (int r = 0; r < 8; ++r) {
        const int row = m0 + wm * 32 + i * 16 + r + 8 * hi;
        const int col = n0 + wn * 64 + j * 16 + ln;
        const size_t idx = (size_t)row * N + col;
        const float v = acc[i][j][r];
        if (EPI == 0) {
          C[idx] = v;
        } else if (EPI == 1) {
          C[idx] = v + res[idx];
        } else {  // EPI == 2
          const float s = rowscale ? rowscale[(size_t)row * rsStride] : 1.f;
          C[idx] = C[idx] + s * v;
        }
      }
}

// ---------------------------------------------------------------------------
// f32 [K][N] -> bf16 [N][K] (transpose + convert) so GEMM B-frags read
// contiguous K per output column.
// ---------------------------------------------------------------------------
__global__ void convT_kernel(const float* __restrict__ w, __bf16* __restrict__ o,
                             int K, int N) {
  const size_t i = (size_t)blockIdx.x * 256 + threadIdx.x;
  if (i < (size_t)K * N) {
    const int k = (int)(i / N);
    const int n = (int)(i % N);
    o[(size_t)n * K + k] = (__bf16)w[i];
  }
}

// ---------------------------------------------------------------------------
// Row RMSNorm (f32 in, bf16 out). One 256-thread block per token.
// ---------------------------------------------------------------------------
__global__ __launch_bounds__(256) void rmsnorm_bf16_kernel(
    const float* __restrict__ x, const float* __restrict__ w,
    __bf16* __restrict__ o, int D) {
  const int t = blockIdx.x, tid = threadIdx.x;
  const float* xr = x + (size_t)t * D;
  float ss = 0.f;
  for (int i = tid; i < D; i += 256) { const float v = xr[i]; ss += v * v; }
  for (int m = 1; m <= 16; m <<= 1) ss += __shfl_xor(ss, m, 32);
  __shared__ float red[8];
  if ((tid & 31) == 0) red[tid >> 5] = ss;
  __syncthreads();
  float tot = 0.f;
#pragma unroll
  for (int i = 0; i < 8; ++i) tot += red[i];
  const float r = rsqrtf(tot / (float)D + 1e-6f);
  __bf16* orow = o + (size_t)t * D;
  for (int i = tid; i < D; i += 256) orow[i] = (__bf16)(xr[i] * r * w[i]);
}

// ---------------------------------------------------------------------------
// Per-head q/k RMSNorm + RoPE + pack into WMMA-friendly layouts.
//   Qb[h][s][64], Kb[kv][s][64] (row-major rows),  Vt[kv][d][s] (transposed)
// One wave per (token, head); lane holds dims d and d+32 (the RoPE pair).
// ---------------------------------------------------------------------------
__global__ __launch_bounds__(128) void rope_pack_kernel(
    const float* __restrict__ qkv, const float* __restrict__ cosb,
    const float* __restrict__ sinb, const float* __restrict__ qw,
    const float* __restrict__ kw, __bf16* __restrict__ Qb,
    __bf16* __restrict__ Kb, __bf16* __restrict__ Vt, int S) {
  const int lane = threadIdx.x & 31, wave = threadIdx.x >> 5;
  const int pair = blockIdx.x * 4 + wave;
  const int s = pair / 24, hh = pair % 24;
  const float* row = qkv + (size_t)s * 1536;
  const int d0 = lane, d1 = lane + 32;
  if (hh < 20) {  // q heads 0..15, k heads 16..19
    const bool isq = hh < 16;
    const int base = isq ? hh * 64 : 1024 + (hh - 16) * 64;
    const float x0 = row[base + d0], x1 = row[base + d1];
    float ss = x0 * x0 + x1 * x1;
    for (int m = 1; m <= 16; m <<= 1) ss += __shfl_xor(ss, m, 32);
    const float r = rsqrtf(ss * (1.f / 64.f) + 1e-6f);
    const float* nw = isq ? qw : kw;
    const float n0 = x0 * r * nw[d0], n1 = x1 * r * nw[d1];
    const float c0 = cosb[(size_t)s * 64 + d0], c1 = cosb[(size_t)s * 64 + d1];
    const float s0 = sinb[(size_t)s * 64 + d0], s1 = sinb[(size_t)s * 64 + d1];
    const float y0 = n0 * c0 - n1 * s0;  // rotate_half: first half gets -x2*sin
    const float y1 = n1 * c1 + n0 * s1;
    if (isq) {
      __bf16* q = Qb + ((size_t)hh * S + s) * 64;
      q[d0] = (__bf16)y0; q[d1] = (__bf16)y1;
    } else {
      __bf16* k = Kb + ((size_t)(hh - 16) * S + s) * 64;
      k[d0] = (__bf16)y0; k[d1] = (__bf16)y1;
    }
  } else {  // v heads 20..23: plain convert, transposed store
    const int kvh = hh - 20;
    const int base = 1280 + kvh * 64;
    Vt[((size_t)kvh * 64 + d0) * S + s] = (__bf16)row[base + d0];
    Vt[((size_t)kvh * 64 + d1) * S + s] = (__bf16)row[base + d1];
  }
}

// ---------------------------------------------------------------------------
// Flash attention, block-causal (BLK=32). Grid = (S/64, H), 4 waves/block,
// each wave owns 16 query rows. Because query-wave tiles never straddle a
// 32-block, the mask reduces to a per-wave key limit kend=(qblk+1)*32 and
// whole 32-key tiles: no element masking. QK^T and P*V both on WMMA bf16.
// ---------------------------------------------------------------------------
__global__ __launch_bounds__(128) void attn_kernel(
    const __bf16* __restrict__ Qb, const __bf16* __restrict__ Kb,
    const __bf16* __restrict__ Vt, __bf16* __restrict__ O, int S) {
  const int h = blockIdx.y, qt = blockIdx.x;
  const int lane = threadIdx.x & 31, wave = threadIdx.x >> 5;
  const int hi = lane >> 4, ln = lane & 15;
  const int q0 = qt * 64 + wave * 16;
  const int kvh = h >> 2;  // rep = H/KV = 4
  const __bf16* Qh = Qb + (size_t)h * S * 64;
  const __bf16* Kh = Kb + (size_t)kvh * S * 64;
  const __bf16* Vh = Vt + (size_t)kvh * 64 * S;

  // Q fragments: 16 rows x HD=64 -> two K=32 A-frags, loaded straight from global
  ABFrag qf[2];
  const int qrow = q0 + ln;
#pragma unroll
  for (int kk = 0; kk < 2; ++kk) {
    qf[kk].h[0] = *(const v8bf*)&Qh[(size_t)qrow * 64 + kk * 32 + 8 * hi];
    qf[kk].h[1] = *(const v8bf*)&Qh[(size_t)qrow * 64 + kk * 32 + 16 + 8 * hi];
  }

  float m_i[8], l_i[8];
  v8f accf[4];
#pragma unroll
  for (int r = 0; r < 8; ++r) { m_i[r] = -1e30f; l_i[r] = 0.f; }
#pragma unroll
  for (int j = 0; j < 4; ++j) accf[j] = zero_v8f();

  __shared__ __align__(16) __bf16 Ps[4][16][40];  // per-wave P staging

  const int kend = ((q0 >> 5) + 1) << 5;  // (qblk+1)*32
  for (int kt = 0; kt < kend; kt += 32) {
    // ---- scores: S(16x32) = Q(16x64) @ K^T(64x32), two 16-col frags
    v8f sf[2];
#pragma unroll
    for (int j = 0; j < 2; ++j) {
      sf[j] = zero_v8f();
      const int key = kt + j * 16 + ln;  // B lane = key column
#pragma unroll
      for (int kk = 0; kk < 2; ++kk) {
        ABFrag bf_;
        bf_.h[0] = *(const v8bf*)&Kh[(size_t)key * 64 + kk * 32 + 8 * hi];
        bf_.h[1] = *(const v8bf*)&Kh[(size_t)key * 64 + kk * 32 + 16 + 8 * hi];
        sf[j] = wmma_bf16(qf[kk], bf_, sf[j]);
      }
    }
    // ---- online softmax over the 32 new keys (row = r + 8*hi)
#pragma unroll
    for (int r = 0; r < 8; ++r) {
      const float s0 = sf[0][r] * 0.125f;  // HD^-0.5
      const float s1 = sf[1][r] * 0.125f;
      float mx = fmaxf(s0, s1);
      for (int m = 1; m <= 8; m <<= 1) mx = fmaxf(mx, __shfl_xor(mx, m, 32));
      const float mnew = fmaxf(m_i[r], mx);
      const float p0 = __expf(s0 - mnew);
      const float p1 = __expf(s1 - mnew);
      float rs = p0 + p1;
      for (int m = 1; m <= 8; m <<= 1) rs += __shfl_xor(rs, m, 32);
      const float alpha = __expf(m_i[r] - mnew);
      l_i[r] = l_i[r] * alpha + rs;
      m_i[r] = mnew;
#pragma unroll
      for (int j = 0; j < 4; ++j) accf[j][r] = accf[j][r] * alpha;
      Ps[wave][r + 8 * hi][ln]      = (__bf16)p0;
      Ps[wave][r + 8 * hi][16 + ln] = (__bf16)p1;
    }
    // ---- O += P(16x32) @ V(32x64); reload P in A-fragment layout
    ABFrag pf;
    pf.h[0] = *(const v8bf*)&Ps[wave][ln][8 * hi];
    pf.h[1] = *(const v8bf*)&Ps[wave][ln][16 + 8 * hi];
#pragma unroll
    for (int j = 0; j < 4; ++j) {
      ABFrag vf;
      const int hd = j * 16 + ln;  // B lane = output (HD) column
      vf.h[0] = *(const v8bf*)&Vh[(size_t)hd * S + kt + 8 * hi];
      vf.h[1] = *(const v8bf*)&Vh[(size_t)hd * S + kt + 16 + 8 * hi];
      accf[j] = wmma_bf16(pf, vf, accf[j]);
    }
  }

  // ---- normalize + store bf16 into [S][H*HD] for the w_o GEMM
#pragma unroll
  for (int j = 0; j < 4; ++j)
#pragma unroll
    for (int r = 0; r < 8; ++r) {
      const int row = q0 + r + 8 * hi;
      const int col = h * 64 + j * 16 + ln;
      O[(size_t)row * 1024 + col] = (__bf16)(accf[j][r] / l_i[r]);
    }
}

// ---------------------------------------------------------------------------
// Router: logits = t @ router_w (1024x8), softmax, top-2, renormalize.
// One wave per token.
// ---------------------------------------------------------------------------
__global__ __launch_bounds__(128) void router_kernel(
    const __bf16* __restrict__ t, const float* __restrict__ rw,
    float* __restrict__ wtok, int S) {
  const int lane = threadIdx.x & 31, wave = threadIdx.x >> 5;
  const int tok = blockIdx.x * 4 + wave;
  if (tok >= S) return;
  float acc[8];
#pragma unroll
  for (int e = 0; e < 8; ++e) acc[e] = 0.f;
  for (int d = lane; d < 1024; d += 32) {
    const float x = (float)t[(size_t)tok * 1024 + d];
#pragma unroll
    for (int e = 0; e < 8; ++e) acc[e] += x * rw[(size_t)d * 8 + e];
  }
#pragma unroll
  for (int e = 0; e < 8; ++e)
    for (int m = 1; m <= 16; m <<= 1) acc[e] += __shfl_xor(acc[e], m, 32);
  // full softmax (all lanes redundantly)
  float mx = acc[0];
#pragma unroll
  for (int e = 1; e < 8; ++e) mx = fmaxf(mx, acc[e]);
  float p[8], se = 0.f;
#pragma unroll
  for (int e = 0; e < 8; ++e) { p[e] = __expf(acc[e] - mx); se += p[e]; }
#pragma unroll
  for (int e = 0; e < 8; ++e) p[e] /= se;
  int i1 = 0; float v1 = p[0];
#pragma unroll
  for (int e = 1; e < 8; ++e) if (p[e] > v1) { v1 = p[e]; i1 = e; }
  int i2 = -1; float v2 = -1.f;
#pragma unroll
  for (int e = 0; e < 8; ++e) if (e != i1 && p[e] > v2) { v2 = p[e]; i2 = e; }
  const float ws = v1 + v2;
  if (lane == 0) {
#pragma unroll
    for (int e = 0; e < 8; ++e) wtok[(size_t)tok * 8 + e] = 0.f;
    wtok[(size_t)tok * 8 + i1] = v1 / ws;
    wtok[(size_t)tok * 8 + i2] = v2 / ws;
  }
}

// ---------------------------------------------------------------------------
// h = silu(g) * u  (f32,f32 -> bf16)
// ---------------------------------------------------------------------------
__global__ void silu_mul_kernel(const float* __restrict__ g,
                                const float* __restrict__ u,
                                __bf16* __restrict__ h, size_t n) {
  const size_t i = (size_t)blockIdx.x * 256 + threadIdx.x;
  if (i < n) {
    const float x = g[i];
    const float s = x / (1.f + __expf(-x));
    h[i] = (__bf16)(s * u[i]);
  }
}

// out = a + b
__global__ void add_kernel(const float* __restrict__ a,
                           const float* __restrict__ b,
                           float* __restrict__ o, size_t n) {
  const size_t i = (size_t)blockIdx.x * 256 + threadIdx.x;
  if (i < n) o[i] = a[i] + b[i];
}

// ---------------------------------------------------------------------------
// Host orchestration (graph-capture safe: only kernel launches on stream)
// ---------------------------------------------------------------------------
extern "C" void kernel_launch(void* const* d_in, const int* in_sizes, int n_in,
                              void* d_out, int out_size, void* d_ws,
                              size_t ws_size, hipStream_t stream) {
  (void)n_in; (void)out_size;
  const int D = 1024, NQKV = 1536, I = 1024, E = 8, HD = 64, H = 16, KV = 4;
  const int S = in_sizes[0] / D;  // B=1

  const float* hidden  = (const float*)d_in[0];
  const float* cosb    = (const float*)d_in[1];
  const float* sinb    = (const float*)d_in[2];
  const float* w_qkv   = (const float*)d_in[3];
  const float* w_o     = (const float*)d_in[4];
  const float* qnw     = (const float*)d_in[5];
  const float* knw     = (const float*)d_in[6];
  const float* inw     = (const float*)d_in[7];
  const float* postw   = (const float*)d_in[8];
  const float* rw      = (const float*)d_in[9];
  const float* w_gate  = (const float*)d_in[10];
  const float* w_up    = (const float*)d_in[11];
  const float* w_down  = (const float*)d_in[12];
  const float* ws_gate = (const float*)d_in[13];
  const float* ws_up   = (const float*)d_in[14];
  const float* ws_down = (const float*)d_in[15];
  float* out = (float*)d_out;

  char* base = (char*)d_ws;
  size_t off = 0;
  auto alloc = [&](size_t bytes) -> char* {
    char* p = base + off;
    off += (bytes + 255) & ~(size_t)255;
    return p;
  };
  __bf16* wqkvT = (__bf16*)alloc((size_t)NQKV * D * 2);
  __bf16* woT   = (__bf16*)alloc((size_t)D * D * 2);
  __bf16* wsgT  = (__bf16*)alloc((size_t)I * D * 2);
  __bf16* wsuT  = (__bf16*)alloc((size_t)I * D * 2);
  __bf16* wsdT  = (__bf16*)alloc((size_t)D * I * 2);
  __bf16* wgT   = (__bf16*)alloc((size_t)I * D * 2);  // reused per expert
  __bf16* wuT   = (__bf16*)alloc((size_t)I * D * 2);
  __bf16* wdT   = (__bf16*)alloc((size_t)D * I * 2);
  __bf16* hA    = (__bf16*)alloc((size_t)S * D * 2);  // norm out / silu out
  float*  qkvF  = (float*) alloc((size_t)S * NQKV * 4);
  __bf16* Qb    = (__bf16*)alloc((size_t)H * S * HD * 2);
  __bf16* Kb    = (__bf16*)alloc((size_t)KV * S * HD * 2);
  __bf16* Vt    = (__bf16*)alloc((size_t)KV * HD * S * 2);
  __bf16* attnO = (__bf16*)alloc((size_t)S * D * 2);
  float*  xF    = (float*) alloc((size_t)S * D * 4);
  __bf16* tB    = (__bf16*)alloc((size_t)S * D * 2);
  float*  wtok  = (float*) alloc((size_t)S * E * 4);
  float*  gbuf  = (float*) alloc((size_t)S * I * 4);
  float*  ubuf  = (float*) alloc((size_t)S * I * 4);
  float*  macc  = (float*) alloc((size_t)S * D * 4);
  if (off > ws_size) return;

  auto cgrid = [](size_t n) { return dim3((unsigned)((n + 255) / 256)); };

  // ---- weight conversion (f32 KxN -> bf16 NxK) ----
  convT_kernel<<<cgrid((size_t)D * NQKV), 256, 0, stream>>>(w_qkv, wqkvT, D, NQKV);
  convT_kernel<<<cgrid((size_t)D * D), 256, 0, stream>>>(w_o, woT, D, D);
  convT_kernel<<<cgrid((size_t)D * I), 256, 0, stream>>>(ws_gate, wsgT, D, I);
  convT_kernel<<<cgrid((size_t)D * I), 256, 0, stream>>>(ws_up, wsuT, D, I);
  convT_kernel<<<cgrid((size_t)I * D), 256, 0, stream>>>(ws_down, wsdT, I, D);

  // ---- attention path ----
  rmsnorm_bf16_kernel<<<S, 256, 0, stream>>>(hidden, inw, hA, D);
  gemm_bf16_wmma<0><<<dim3(NQKV / 128, S / 128), 256, 0, stream>>>(
      hA, wqkvT, qkvF, S, NQKV, D, nullptr, nullptr, 0);
  rope_pack_kernel<<<(S * 24) / 4, 128, 0, stream>>>(qkvF, cosb, sinb, qnw, knw,
                                                     Qb, Kb, Vt, S);
  attn_kernel<<<dim3(S / 64, H), 128, 0, stream>>>(Qb, Kb, Vt, attnO, S);
  gemm_bf16_wmma<1><<<dim3(D / 128, S / 128), 256, 0, stream>>>(
      attnO, woT, xF, S, D, D, hidden, nullptr, 0);  // x = hidden + attn@w_o

  // ---- MoE path ----
  rmsnorm_bf16_kernel<<<S, 256, 0, stream>>>(xF, postw, tB, D);
  router_kernel<<<S / 4, 128, 0, stream>>>(tB, rw, wtok, S);

  // shared expert (writes macc, overwriting poisoned workspace)
  gemm_bf16_wmma<0><<<dim3(I / 128, S / 128), 256, 0, stream>>>(
      tB, wsgT, gbuf, S, I, D, nullptr, nullptr, 0);
  gemm_bf16_wmma<0><<<dim3(I / 128, S / 128), 256, 0, stream>>>(
      tB, wsuT, ubuf, S, I, D, nullptr, nullptr, 0);
  silu_mul_kernel<<<cgrid((size_t)S * I), 256, 0, stream>>>(gbuf, ubuf, hA,
                                                            (size_t)S * I);
  gemm_bf16_wmma<0><<<dim3(D / 128, S / 128), 256, 0, stream>>>(
      hA, wsdT, macc, S, D, I, nullptr, nullptr, 0);

  // routed experts: dense per-expert compute, weighted accumulate (matches
  // the reference, which also evaluates every expert and weights by w_tok)
  for (int e = 0; e < E; ++e) {
    convT_kernel<<<cgrid((size_t)D * I), 256, 0, stream>>>(
        w_gate + (size_t)e * D * I, wgT, D, I);
    convT_kernel<<<cgrid((size_t)D * I), 256, 0, stream>>>(
        w_up + (size_t)e * D * I, wuT, D, I);
    convT_kernel<<<cgrid((size_t)I * D), 256, 0, stream>>>(
        w_down + (size_t)e * I * D, wdT, I, D);
    gemm_bf16_wmma<0><<<dim3(I / 128, S / 128), 256, 0, stream>>>(
        tB, wgT, gbuf, S, I, D, nullptr, nullptr, 0);
    gemm_bf16_wmma<0><<<dim3(I / 128, S / 128), 256, 0, stream>>>(
        tB, wuT, ubuf, S, I, D, nullptr, nullptr, 0);
    silu_mul_kernel<<<cgrid((size_t)S * I), 256, 0, stream>>>(gbuf, ubuf, hA,
                                                              (size_t)S * I);
    gemm_bf16_wmma<2><<<dim3(D / 128, S / 128), 256, 0, stream>>>(
        hA, wdT, macc, S, D, I, nullptr, wtok + e, E);
  }

  // ---- final residual: out = x + (moe + shared) ----
  add_kernel<<<cgrid((size_t)S * D), 256, 0, stream>>>(xF, macc, out,
                                                       (size_t)S * D);
}
